// MultiHeadAttention_89249420411636
// MI455X (gfx1250) — matrix operand order repaired and checked by reading, not compiled, not run on hardware
//
#include <hip/hip_runtime.h>

typedef __bf16 bf16_t;
typedef __attribute__((ext_vector_type(16))) __bf16 v16bf;
typedef __attribute__((ext_vector_type(8)))  __bf16 v8bf;
typedef __attribute__((ext_vector_type(8)))  float  v8f;

#define WMMA_BF16(a, b, c) \
  __builtin_amdgcn_wmma_f32_16x16x32_bf16(false, (a), false, (b), (short)0, (c), false, false)

// ---------------------------------------------------------------------------
// Fragment loaders (CDNA5 wave32 WMMA 16x16x32 bf16 layouts, CDNA5 ISA)
// A (16x32, MxK): lane l: m = l&15; elems 0..7 -> k = (l>>4)*8 + 0..7,
//                 elems 8..15 -> k = 16 + (l>>4)*8 + 0..7
// B (32x16, KxN) from B^T row-major [N,K]: lane l: n = l&15; elems 0..7 -> k = (l>>4)*8 + 0..7,
//                 elems 8..15 -> k = 16 + (l>>4)*8 + 0..7
// C/D (16x16 f32): lane l: n = l&15; vgpr r -> m = r + 8*(l>>4)
// ---------------------------------------------------------------------------
__device__ __forceinline__ v16bf load_a_bf16(const bf16_t* tile, int ld, int lane) {
  int m = lane & 15, g = lane >> 4;
  const bf16_t* p = tile + (size_t)m * ld + g * 8;
  v8bf lo = *(const v8bf*)(p);
  v8bf hi = *(const v8bf*)(p + 16);
  return __builtin_shufflevector(lo, hi, 0,1,2,3,4,5,6,7,8,9,10,11,12,13,14,15);
}

__device__ __forceinline__ v16bf load_a_f32(const float* tile, int ld, int lane) {
  int m = lane & 15, g = lane >> 4;
  const float* p = tile + (size_t)m * ld + g * 8;
  v16bf f;
#pragma unroll
  for (int i = 0; i < 8; ++i) { f[i] = (__bf16)p[i]; f[i + 8] = (__bf16)p[i + 16]; }
  return f;
}

__device__ __forceinline__ v16bf load_b_bf16(const bf16_t* tileT, int ld, int lane) {
  int n = lane & 15, g = lane >> 4;
  const bf16_t* p = tileT + (size_t)n * ld + g * 8;
  v8bf lo = *(const v8bf*)(p);
  v8bf hi = *(const v8bf*)(p + 16);
  return __builtin_shufflevector(lo, hi, 0,1,2,3,4,5,6,7,8,9,10,11,12,13,14,15);
}

// ---------------------------------------------------------------------------
// Constants
// ---------------------------------------------------------------------------
#define BB     8
#define SS     2048
#define DIN    64
#define DEMB   128
#define NHEAD  8
#define FF     1024          // NHEAD * DEMB
#define MTOT   16384         // BB * SS
#define INV_TEMP 0.08838834764831845f   // 1/sqrt(128)
#define NEGBIG  -1.0e30f

// ---------------------------------------------------------------------------
// K0: weight fp32 -> bf16
// ---------------------------------------------------------------------------
__global__ void k_convert(const float* w_in, const float* w_mh, const float* w_fc,
                          bf16_t* wi, bf16_t* wm, bf16_t* wf) {
  int i = blockIdx.x * 256 + threadIdx.x;
  if (i < DEMB * DIN) wi[i] = (__bf16)w_in[i];
  if (i < FF * DEMB) { wm[i] = (__bf16)w_mh[i]; wf[i] = (__bf16)w_fc[i]; }
}

// ---------------------------------------------------------------------------
// K1: shared embedding  Xe = X @ W_in^T   ([16384,64] x [64,128])
// grid (8 ntiles, 128, 3), block 256 (8 waves, one 16x16 tile each)
// ---------------------------------------------------------------------------
__global__ __launch_bounds__(256) void k_embed(const float* Q, const float* K, const float* V,
                                               const bf16_t* w_in, float* Qe32,
                                               bf16_t* Qeb, bf16_t* Keb, bf16_t* Veb) {
  int lane = threadIdx.x & 31, wave = threadIdx.x >> 5;
  int mt = blockIdx.y * 8 + wave;
  int nt = blockIdx.x;
  const float* X = (blockIdx.z == 0) ? Q : (blockIdx.z == 1) ? K : V;
  bf16_t* O = (blockIdx.z == 0) ? Qeb : (blockIdx.z == 1) ? Keb : Veb;
  const float* atile = X + (size_t)mt * 16 * DIN;
  const bf16_t* btile = w_in + (size_t)nt * 16 * DIN;
  v8f c = {};
#pragma unroll
  for (int ks = 0; ks < DIN; ks += 32) {
    v16bf a = load_a_f32(atile + ks, DIN, lane);
    v16bf b = load_b_bf16(btile + ks, DIN, lane);
    c = WMMA_BF16(a, b, c);
  }
  int n = lane & 15, hi = lane >> 4;
  size_t row0 = (size_t)mt * 16;
  int col = nt * 16 + n;
  if (blockIdx.z == 0) {   // uniform branch, keeps store loops clean
#pragma unroll
    for (int r = 0; r < 8; ++r)
      Qe32[(row0 + r + 8 * hi) * DEMB + col] = c[r];
  }
#pragma unroll
  for (int r = 0; r < 8; ++r)
    O[(row0 + r + 8 * hi) * DEMB + col] = (__bf16)c[r];
}

// ---------------------------------------------------------------------------
// K2: multi-head projection  ([16384,128] x [128,1024])
// z=0 -> q (scaled 1/TEMP) [B,H,S,128]; z=1 -> k [B,H,S,128]; z=2 -> v^T [B,H,128,S]
// grid (64, 128, 3), block 256
// ---------------------------------------------------------------------------
__global__ __launch_bounds__(256) void k_project(const bf16_t* Qeb, const bf16_t* Keb,
                                                 const bf16_t* Veb, const bf16_t* w_mh,
                                                 bf16_t* qb, bf16_t* kb, bf16_t* vtb) {
  int lane = threadIdx.x & 31, wave = threadIdx.x >> 5;
  int mt = blockIdx.y * 8 + wave;
  int nt = blockIdx.x;
  int z = blockIdx.z;
  const bf16_t* X = (z == 0) ? Qeb : (z == 1) ? Keb : Veb;
  const bf16_t* atile = X + (size_t)mt * 16 * DEMB;
  const bf16_t* btile = w_mh + (size_t)nt * 16 * DEMB;
  v8f c = {};
#pragma unroll
  for (int ks = 0; ks < DEMB; ks += 32) {
    v16bf a = load_a_bf16(atile + ks, DEMB, lane);
    v16bf b = load_b_bf16(btile + ks, DEMB, lane);
    c = WMMA_BF16(a, b, c);
  }
  float scale = (z == 0) ? INV_TEMP : 1.0f;
  int n = lane & 15, hi = lane >> 4;
  int f = nt * 16 + n;
  int h = f >> 7, d = f & 127;
#pragma unroll
  for (int r = 0; r < 8; ++r) {
    int m = r + 8 * hi;
    size_t gm = (size_t)mt * 16 + m;
    int b_ = (int)(gm >> 11);
    int s = (int)(gm & 2047);
    size_t bh = (size_t)b_ * NHEAD + h;
    float val = c[r] * scale;
    if (z == 0)      qb[(bh * SS + s) * DEMB + d] = (__bf16)val;
    else if (z == 1) kb[(bh * SS + s) * DEMB + d] = (__bf16)val;
    else             vtb[(bh * DEMB + d) * SS + s] = (__bf16)val;
  }
}

// ---------------------------------------------------------------------------
// K3: flash attention per (b,h). Block = 128 q rows (8 waves x 16 rows),
// Sk tiled by 64, online softmax via per-wave LDS, prefetch of next K/V tiles.
// Output stored in the reference's scrambled reshape layout as bf16.
// grid (16, 64), block 256
// ---------------------------------------------------------------------------
__global__ __launch_bounds__(256) void k_attn(const bf16_t* qb, const bf16_t* kb,
                                              const bf16_t* vtb, const float* QMask,
                                              bf16_t* o2b) {
  __shared__ float  Sb[8][16 * 64];
  __shared__ bf16_t Pb[8][16 * 64];
  __shared__ float  Stats[8][16];

  int lane = threadIdx.x & 31, wave = threadIdx.x >> 5;
  int bh = blockIdx.y;
  int b_ = bh >> 3, h = bh & 7;
  int q0 = blockIdx.x * 128 + wave * 16;

  const bf16_t* qbase = qb + ((size_t)bh * SS + q0) * DEMB;
  const bf16_t* kbase = kb + (size_t)bh * SS * DEMB;
  const bf16_t* vbase = vtb + (size_t)bh * DEMB * SS;
  const float* mask = QMask + (size_t)b_ * SS;

  int n = lane & 15, hi = lane >> 4;
  float* Sw = &Sb[wave][0];
  bf16_t* Pw = &Pb[wave][0];
  float* St = &Stats[wave][0];

  // preload the wave's 16x128 q strip as 4 A-fragments
  v16bf qf[4];
#pragma unroll
  for (int s2 = 0; s2 < 4; ++s2) qf[s2] = load_a_bf16(qbase + s2 * 32, DEMB, lane);

  v8f O[8];
#pragma unroll
  for (int t = 0; t < 8; ++t) O[t] = (v8f){};
  float runM = -3.0e38f, runL = 0.0f;   // valid per lane for row (lane&15)

  for (int kt = 0; kt < SS; kt += 64) {
    // ---- prefetch next K/V tiles into cache (global_prefetch_b8) ----
    if (kt + 64 < SS) {
      const bf16_t* nk = kbase + (size_t)(kt + 64) * DEMB;
      const bf16_t* nv = vbase + (kt + 64);
#pragma unroll
      for (int i = 0; i < 4; ++i) {
        __builtin_prefetch(nk + ((size_t)lane + 32 * i) * 64, 0, 1);
        __builtin_prefetch(nv + ((size_t)lane + 32 * i) * SS, 0, 1);
      }
    }
    // ---- GEMM1: S = q * k^T for 4 column tiles, +mask, spill to LDS ----
#pragma unroll
    for (int ct = 0; ct < 4; ++ct) {
      v8f sc = {};
      const bf16_t* kt0 = kbase + (size_t)(kt + ct * 16) * DEMB;
#pragma unroll
      for (int s2 = 0; s2 < 4; ++s2) {
        v16bf bf = load_b_bf16(kt0 + s2 * 32, DEMB, lane);
        sc = WMMA_BF16(qf[s2], bf, sc);
      }
      float mv = mask[kt + ct * 16 + n];
      float addv = (mv == 0.0f) ? NEGBIG : 0.0f;
#pragma unroll
      for (int r = 0; r < 8; ++r)
        Sw[(r + 8 * hi) * 64 + ct * 16 + n] = sc[r] + addv;
    }
    // ---- online softmax: lane owns row n, column half hi (32 cols) ----
    float vals[32];
    float tmax = -3.0e38f;
#pragma unroll
    for (int j = 0; j < 32; ++j) {
      vals[j] = Sw[n * 64 + hi * 32 + j];
      tmax = fmaxf(tmax, vals[j]);
    }
    tmax = fmaxf(tmax, __shfl_xor(tmax, 16, 32));
    float newM = fmaxf(runM, tmax);
    float alpha = __expf(runM - newM);
    float psum = 0.0f;
#pragma unroll
    for (int j = 0; j < 32; ++j) {
      float p = __expf(vals[j] - newM);
      psum += p;
      Pw[n * 64 + hi * 32 + j] = (__bf16)p;
    }
    psum += __shfl_xor(psum, 16, 32);
    runL = runL * alpha + psum;
    runM = newM;
    if (hi == 0) St[n] = alpha;
    // ---- GEMM2: O = O*alpha + P * V  (8 d-tiles x 2 K-slabs) ----
#pragma unroll
    for (int t = 0; t < 8; ++t) {
      v8f o = O[t];
#pragma unroll
      for (int r = 0; r < 8; ++r) o[r] *= St[r + 8 * hi];
#pragma unroll
      for (int s2 = 0; s2 < 2; ++s2) {
        v16bf pf = load_a_bf16(Pw + s2 * 32, 64, lane);
        v16bf vf = load_b_bf16(vbase + (size_t)(t * 16) * SS + kt + s2 * 32, SS, lane);
        o = WMMA_BF16(pf, vf, o);
      }
      O[t] = o;
    }
  }
  // ---- finalize: divide by row sums, store to scrambled concat layout ----
  if (hi == 0) St[n] = runL;
  float linv[8];
#pragma unroll
  for (int r = 0; r < 8; ++r) linv[r] = 1.0f / St[r + 8 * hi];
#pragma unroll
  for (int t = 0; t < 8; ++t) {
#pragma unroll
    for (int r = 0; r < 8; ++r) {
      int m = r + 8 * hi;
      float val = O[t][r] * linv[r];
      int s = q0 + m;
      int d = t * 16 + n;
      size_t sp = (size_t)h * 256 + (s >> 3);
      size_t fidx = (size_t)(s & 7) * 128 + d;
      o2b[((size_t)b_ * SS + sp) * FF + fidx] = (__bf16)val;
    }
  }
}

// ---------------------------------------------------------------------------
// K4: out = o2 @ W_fc^T + residual, then LayerNorm.  ([16384,1024]x[1024,128])
// The 16x1024 A strip (32KB, contiguous) is shared by all 8 waves: stage it
// into LDS once with global_load_async_to_lds_b128 (ASYNCcnt), then feed
// WMMA A-fragments from LDS. grid (1024), block 256.
// ---------------------------------------------------------------------------
__global__ __launch_bounds__(256) void k_out(const bf16_t* o2b, const bf16_t* w_fc,
                                             const float* Qe32, const float* ln_g,
                                             const float* ln_b, float* out) {
  __shared__ bf16_t Ab[16 * FF];   // 32 KB A strip
  __shared__ float  Ob[16][128];   // 8 KB LN staging
  int lane = threadIdx.x & 31, wave = threadIdx.x >> 5;
  int mt = blockIdx.x;
  int nt = wave;

  // ---- async stage of contiguous 32KB A strip: 2048 x 16B chunks ----
  {
    typedef __attribute__((address_space(3))) char lds_char;
    unsigned lbase = (unsigned)(unsigned long long)(lds_char*)&Ab[0];
    unsigned long long gbase = (unsigned long long)(const void*)(o2b + (size_t)mt * 16 * FF);
#pragma unroll
    for (int i = 0; i < 8; ++i) {
      unsigned off = (threadIdx.x + i * 256) * 16;
      asm volatile("global_load_async_to_lds_b128 %0, %1, %2"
                   :: "v"(lbase + off), "v"(off), "s"(gbase) : "memory");
    }
    asm volatile("s_wait_asynccnt 0" ::: "memory");
  }
  __syncthreads();

  const bf16_t* btile = w_fc + (size_t)nt * 16 * FF;
  v8f c = {};
  for (int ks = 0; ks < FF; ks += 32) {
    v16bf a = load_a_bf16(Ab + ks, FF, lane);   // ds_load from LDS
    v16bf b = load_b_bf16(btile + ks, FF, lane);
    c = WMMA_BF16(a, b, c);
  }
  int n = lane & 15, hi = lane >> 4;
#pragma unroll
  for (int r = 0; r < 8; ++r) {
    int m = r + 8 * hi;
    int col = nt * 16 + n;
    Ob[m][col] = c[r] + Qe32[((size_t)mt * 16 + m) * DEMB + col];
  }
  __syncthreads();
  int m = threadIdx.x >> 4;
  int c0 = (threadIdx.x & 15) * 8;
  float x[8];
  float sum = 0.0f, sq = 0.0f;
#pragma unroll
  for (int i = 0; i < 8; ++i) { x[i] = Ob[m][c0 + i]; sum += x[i]; sq += x[i] * x[i]; }
#pragma unroll
  for (int w = 1; w < 16; w <<= 1) {
    sum += __shfl_xor(sum, w, 32);
    sq  += __shfl_xor(sq,  w, 32);
  }
  float mean = sum * (1.0f / 128.0f);
  float var  = sq * (1.0f / 128.0f) - mean * mean;
  float rstd = rsqrtf(var + 1e-6f);
  size_t row = (size_t)mt * 16 + m;
#pragma unroll
  for (int i = 0; i < 8; ++i)
    out[row * DEMB + c0 + i] = (x[i] - mean) * rstd * ln_g[c0 + i] + ln_b[c0 + i];
}

// ---------------------------------------------------------------------------
// Launcher
// ---------------------------------------------------------------------------
extern "C" void kernel_launch(void* const* d_in, const int* in_sizes, int n_in,
                              void* d_out, int out_size, void* d_ws, size_t ws_size,
                              hipStream_t stream) {
  const float* Q     = (const float*)d_in[0];
  const float* K     = (const float*)d_in[1];
  const float* V     = (const float*)d_in[2];
  const float* QMask = (const float*)d_in[3];
  const float* W_in  = (const float*)d_in[4];
  const float* W_mh  = (const float*)d_in[5];
  const float* W_fc  = (const float*)d_in[6];
  const float* ln_g  = (const float*)d_in[7];
  const float* ln_b  = (const float*)d_in[8];
  float* out = (float*)d_out;

  char* ws = (char*)d_ws;
  size_t off = 0;
  auto take = [&](size_t bytes) -> char* {
    char* p = ws + off;
    off += (bytes + 255) & ~(size_t)255;
    return p;
  };
  bf16_t* wi   = (bf16_t*)take((size_t)DEMB * DIN * 2);
  bf16_t* wm   = (bf16_t*)take((size_t)FF * DEMB * 2);
  bf16_t* wf   = (bf16_t*)take((size_t)FF * DEMB * 2);
  float*  Qe32 = (float*) take((size_t)MTOT * DEMB * 4);
  bf16_t* Qeb  = (bf16_t*)take((size_t)MTOT * DEMB * 2);
  bf16_t* Keb  = (bf16_t*)take((size_t)MTOT * DEMB * 2);
  bf16_t* Veb  = (bf16_t*)take((size_t)MTOT * DEMB * 2);
  bf16_t* qbuf = (bf16_t*)take((size_t)BB * NHEAD * SS * DEMB * 2);
  bf16_t* kbuf = (bf16_t*)take((size_t)BB * NHEAD * SS * DEMB * 2);
  bf16_t* vtb  = (bf16_t*)take((size_t)BB * NHEAD * DEMB * SS * 2);
  bf16_t* o2b  = (bf16_t*)take((size_t)BB * SS * FF * 2);

  k_convert<<<dim3(512), dim3(256), 0, stream>>>(W_in, W_mh, W_fc, wi, wm, wf);
  k_embed<<<dim3(8, 128, 3), dim3(256), 0, stream>>>(Q, K, V, wi, Qe32, Qeb, Keb, Veb);
  k_project<<<dim3(64, 128, 3), dim3(256), 0, stream>>>(Qeb, Keb, Veb, wm, qbuf, kbuf, vtb);
  k_attn<<<dim3(16, 64), dim3(256), 0, stream>>>(qbuf, kbuf, vtb, QMask, o2b);
  k_out<<<dim3(1024), dim3(256), 0, stream>>>(o2b, wf, Qe32, ln_g, ln_b, out);
}
